// DeltaNet_31877247271480
// MI455X (gfx1250) — matrix operand order repaired and checked by reading, not compiled, not run on hardware
//
// DeltaNet forward for MI455X (gfx1250, wave32, WMMA bf16 -> f32 accum,
// async global->LDS double-buffered GEMM staging).
#include <hip/hip_runtime.h>
#include <hip/hip_bf16.h>
#include <math.h>

typedef __attribute__((ext_vector_type(16))) __bf16 v16bf;
typedef __attribute__((ext_vector_type(8)))  float  v8f;
typedef __attribute__((ext_vector_type(4)))  unsigned int u32x4;
typedef __attribute__((ext_vector_type(4)))  float  f32x4;
typedef unsigned short u16;

#define DEV __device__ __forceinline__

static constexpr int Bb = 2, Ll = 2048, NHh = 4, DVv = 256;
static constexpr int BLn = Bb * Ll;               // 4096
static constexpr int NCH = Ll / 32;               // 64 chunks
static constexpr int GIN = 1056, MLPH = 2048;

DEV u16 f2bfu(float f) {
  union { float f; unsigned u; } a; a.f = f;
  unsigned r = a.u + 0x7FFFu + ((a.u >> 16) & 1u);
  return (u16)(r >> 16);
}
DEV float bf2f(u16 h) {
  union { unsigned u; float f; } a; a.u = ((unsigned)h) << 16;
  return a.f;
}
DEV u32x4 pack8(const u16* p) {
  u32x4 r;
  r.x = (unsigned)p[0] | ((unsigned)p[1] << 16);
  r.y = (unsigned)p[2] | ((unsigned)p[3] << 16);
  r.z = (unsigned)p[4] | ((unsigned)p[5] << 16);
  r.w = (unsigned)p[6] | ((unsigned)p[7] << 16);
  return r;
}

// A fragment: 16x32 bf16, row-major source (row stride ld elements).
// Lanes 0-15: row m, k={0..7,16..23}; lanes 16-31: k={8..15,24..31}.
DEV v16bf ldfragA(const u16* src, int ld) {
  int lane = threadIdx.x & 31;
  int m = lane & 15, kb = (lane >> 4) * 8;
  union { v16bf v; u32x4 q[2]; } r;
  r.q[0] = *(const u32x4*)(src + (size_t)m * ld + kb);
  r.q[1] = *(const u32x4*)(src + (size_t)m * ld + kb + 16);
  return r.v;
}
// B fragment: 32x16 bf16 from N-major (transposed) source srcT[n][k].
DEV v16bf ldfragBT(const u16* srcT, int ld) {
  int lane = threadIdx.x & 31;
  int n = lane & 15, ko = (lane >> 4) * 8;
  union { v16bf v; u32x4 q[2]; } r;
  r.q[0] = *(const u32x4*)(srcT + (size_t)n * ld + ko);
  r.q[1] = *(const u32x4*)(srcT + (size_t)n * ld + ko + 16);
  return r.v;
}
DEV v8f wmma_bf(v16bf a, v16bf b, v8f c) {
  return __builtin_amdgcn_wmma_f32_16x16x32_bf16(false, a, false, b, (short)0, c, false, false);
}
// D layout: lane holds (m = v + 8*(lane>>4), n = lane&15), v = 0..7.
DEV void stfragD_f32(float* dst, int ld, v8f c) {
  int lane = threadIdx.x & 31;
  int n = lane & 15, mb = (lane >> 4) * 8;
#pragma unroll
  for (int v = 0; v < 8; ++v) dst[(size_t)(mb + v) * ld + n] = c[v];
}

// 16-byte async global->LDS copy (ASYNCcnt-tracked, no VGPR round trip).
DEV void async_cp16(unsigned lds_off, const u16* g) {
  asm volatile("global_load_async_to_lds_b128 %0, %1, off"
               :: "v"(lds_off), "v"(g) : "memory");
}
DEV void wait_async3() { asm volatile("s_wait_asynccnt 0x3" ::: "memory"); }
DEV void wait_async0() { asm volatile("s_wait_asynccnt 0x0" ::: "memory"); }

DEV float siluf(float x) { return x / (1.f + expf(-x)); }
DEV float sigmoidf_(float x) { return 1.f / (1.f + expf(-x)); }
DEV float softplusf_(float x) { return x > 15.f ? x : log1pf(expf(x)); }

// ---------------------------------------------------------------------------
// GEMM: C[M,N](f32) = A[M,K](bf16 row-major) * BT[N,K](bf16, pre-transposed).
// 256 thr = 8 waves; block tile 64x128; wave tile 16x64 (4 accumulators).
// Double-buffered async global->LDS staging overlapped with WMMA.
// ---------------------------------------------------------------------------
__global__ __launch_bounds__(256) void gemm_bf16_async(
    const u16* __restrict__ A, const u16* __restrict__ BT,
    float* __restrict__ C, int M, int N, int K) {
  __shared__ __align__(16) u16 As[2][64][32];
  __shared__ __align__(16) u16 Bs[2][128][32];
  int m0 = blockIdx.y * 64, n0 = blockIdx.x * 128;
  int tid = threadIdx.x, wv = tid >> 5;
  int mt = wv >> 1, ns = (wv & 1) * 4;

  // A: thread copies one 16B chunk (row ra, 8 elems at ca)
  int ra = tid >> 2, ca = (tid & 3) * 8;
  const u16* ga = A + (size_t)(m0 + ra) * K + ca;
  unsigned laA[2] = {(unsigned)(uintptr_t)&As[0][ra][ca],
                     (unsigned)(uintptr_t)&As[1][ra][ca]};
  // BT: thread copies 32B (row rb, 16 elems at cb) as two b128s
  int rb = tid >> 1, cb = (tid & 1) * 16;
  const u16* gb = BT + (size_t)(n0 + rb) * K + cb;
  unsigned laB[2] = {(unsigned)(uintptr_t)&Bs[0][rb][cb],
                     (unsigned)(uintptr_t)&Bs[1][rb][cb]};

  v8f acc[4] = {{}, {}, {}, {}};
  int nk = K / 32;
  auto issue = [&](int t) {
    const u16* gat = ga + t * 32;
    const u16* gbt = gb + t * 32;
    int bsel = t & 1;
    async_cp16(laA[bsel], gat);
    async_cp16(laB[bsel], gbt);
    async_cp16(laB[bsel] + 16, gbt + 8);
  };
  issue(0);
  for (int t = 0; t < nk; ++t) {
    if (t + 1 < nk) {
      issue(t + 1);
      wait_async3();                 // tile t's 3 loads have landed
    } else {
      wait_async0();
    }
    if (t + 2 < nk) {                // warm L2 for the tile after next
      __builtin_prefetch(ga + (t + 2) * 32, 0, 1);
      __builtin_prefetch(gb + (t + 2) * 32, 0, 1);
    }
    __syncthreads();
    const u16(*Ab)[32] = As[t & 1];
    const u16(*Bbuf)[32] = Bs[t & 1];
    v16bf a = ldfragA(&Ab[mt * 16][0], 32);
#pragma unroll
    for (int j = 0; j < 4; ++j) {
      v16bf b = ldfragBT(&Bbuf[(ns + j) * 16][0], 32);
      acc[j] = wmma_bf(a, b, acc[j]);
    }
    __syncthreads();
  }
#pragma unroll
  for (int j = 0; j < 4; ++j)
    stfragD_f32(C + (size_t)(m0 + mt * 16) * N + n0 + (ns + j) * 16, N, acc[j]);
}

// Small-N GEMM (N==16): logits = A[M,K] @ BT[16][K]^T. One wave per 16-row tile.
__global__ __launch_bounds__(256) void gemm_n16(const u16* __restrict__ A,
                                                const u16* __restrict__ BT,
                                                float* __restrict__ C, int K) {
  int wv = threadIdx.x >> 5;
  int m0 = (blockIdx.x * 8 + wv) * 16;
  v8f acc = {};
  for (int k0 = 0; k0 < K; k0 += 32) {
    v16bf a = ldfragA(A + (size_t)m0 * K + k0, K);
    v16bf b = ldfragBT(BT + k0, K);
    acc = wmma_bf(a, b, acc);
  }
  stfragD_f32(C + (size_t)m0 * 16, 16, acc);
}

// ---------------------------------------------------------------------------
// Elementwise helpers
// ---------------------------------------------------------------------------
__global__ __launch_bounds__(256) void cvt_bf16(const float* __restrict__ s,
                                                u16* __restrict__ d, int n) {
  int i = blockIdx.x * 256 + threadIdx.x;
  if (i < n) d[i] = f2bfu(s[i]);
}
// src[R][C] f32 -> dst[C][R] bf16 (one-time weight transposes)
__global__ __launch_bounds__(256) void transpose_bf16(const float* __restrict__ s,
                                                      u16* __restrict__ d,
                                                      int R, int Cc) {
  int i = blockIdx.x * 256 + threadIdx.x;
  if (i < R * Cc) { int r = i / Cc, c = i % Cc; d[(size_t)c * R + r] = f2bfu(s[i]); }
}
// causal depthwise conv (cross-correlation, left pad K-1); x,y: [B,L,1024]; w: [1024,K]
template <int KK, bool SILU>
__global__ __launch_bounds__(256) void conv_dw(const float* __restrict__ x,
                                               const float* __restrict__ w,
                                               float* __restrict__ y, int total) {
  int i = blockIdx.x * 256 + threadIdx.x;
  if (i >= total) return;
  int c = i & 1023;
  int bl = i >> 10;
  int l = bl & (Ll - 1);
  int blbase = bl - l;
  float acc = 0.f;
#pragma unroll
  for (int j = 0; j < KK; ++j) {
    int ls = l - (KK - 1) + j;
    if (ls >= 0) acc += w[c * KK + j] * x[((size_t)(blbase + ls) << 10) + c];
  }
  if (SILU) acc = siluf(acc);
  y[i] = acc;
}
__global__ __launch_bounds__(256) void beta_k(const float* __restrict__ x,
                                              const float* __restrict__ Wb,
                                              float* __restrict__ beta) {
  int i = blockIdx.x * 256 + threadIdx.x;     // BLn*4 threads
  int hh = i & 3, row = i >> 2;
  const float* xr = x + ((size_t)row << 10);
  float acc = 0.f;
  for (int j = 0; j < 1024; ++j) acc += xr[j] * Wb[j * 4 + hh];
  beta[i] = sigmoidf_(acc);
}
__global__ __launch_bounds__(256) void gelu_bias(const float* __restrict__ h,
                                                 const float* __restrict__ b1,
                                                 u16* __restrict__ o, int total) {
  int i = blockIdx.x * 256 + threadIdx.x;
  if (i >= total) return;
  float xx = h[i] + b1[i & (MLPH - 1)];
  o[i] = f2bfu(0.5f * xx * (1.f + erff(xx * 0.70710678118f)));
}

// ---------------------------------------------------------------------------
// Delta-rule prep: per (b,h,chunk): l2norm q/k, T = inv(I + tril(kb k^T,-1)),
// u = T (v*beta), w = T (k*beta). WMMA for A, u, w. 512 blocks x 256 thr.
// ---------------------------------------------------------------------------
__global__ __launch_bounds__(256) void delta_prep(
    const float* __restrict__ qc, const float* __restrict__ kc,
    const float* __restrict__ vc, const float* __restrict__ beta,
    u16* __restrict__ qn_g, u16* __restrict__ kn_g, u16* __restrict__ knT_g,
    u16* __restrict__ u_g, u16* __restrict__ w_g) {
  __shared__ __align__(16) u16 knb[32][256];
  __shared__ __align__(16) u16 kbb[32][256];
  __shared__ __align__(16) u16 vtb[256][32];   // (v*beta)^T
  __shared__ __align__(16) u16 kbt[256][32];   // (k*beta)^T
  __shared__ float Af[32][32];
  __shared__ __align__(16) u16 Tb[32][32];
  __shared__ float betas[32];

  int ch = blockIdx.x & 63;
  int h = (blockIdx.x >> 6) & 3;
  int b = blockIdx.x >> 8;
  int bh = b * 4 + h;
  int l0 = ch * 32;
  int tid = threadIdx.x, wv = tid >> 5, lane = tid & 31;

  if (tid < 32) betas[tid] = beta[(size_t)(b * Ll + l0 + tid) * 4 + h];
  __syncthreads();

#pragma unroll
  for (int rr = 0; rr < 4; ++rr) {   // q rows: l2norm -> qn_g (bf16)
    int r = wv * 4 + rr;
    const float* rowp = qc + (size_t)(b * Ll + l0 + r) * 1024 + h * 256;
    f32x4 a0 = *(const f32x4*)(rowp + lane * 8);
    f32x4 a1 = *(const f32x4*)(rowp + lane * 8 + 4);
    float ss = a0.x * a0.x + a0.y * a0.y + a0.z * a0.z + a0.w * a0.w +
               a1.x * a1.x + a1.y * a1.y + a1.z * a1.z + a1.w * a1.w;
#pragma unroll
    for (int m = 16; m; m >>= 1) ss += __shfl_xor(ss, m, 32);
    float sc = rsqrtf(ss + 1e-6f);
    float vals[8] = {a0.x, a0.y, a0.z, a0.w, a1.x, a1.y, a1.z, a1.w};
    u16 pk[8];
#pragma unroll
    for (int j = 0; j < 8; ++j) pk[j] = f2bfu(vals[j] * sc);
    *(u32x4*)(qn_g + (size_t)(bh * Ll + l0 + r) * 256 + lane * 8) = pack8(pk);
  }
#pragma unroll
  for (int rr = 0; rr < 4; ++rr) {   // k rows: kn, kb = kn*beta, kn^T
    int r = wv * 4 + rr;
    const float* rowp = kc + (size_t)(b * Ll + l0 + r) * 1024 + h * 256;
    f32x4 a0 = *(const f32x4*)(rowp + lane * 8);
    f32x4 a1 = *(const f32x4*)(rowp + lane * 8 + 4);
    float ss = a0.x * a0.x + a0.y * a0.y + a0.z * a0.z + a0.w * a0.w +
               a1.x * a1.x + a1.y * a1.y + a1.z * a1.z + a1.w * a1.w;
#pragma unroll
    for (int m = 16; m; m >>= 1) ss += __shfl_xor(ss, m, 32);
    float sc = rsqrtf(ss + 1e-6f);
    float bt = betas[r];
    float vals[8] = {a0.x, a0.y, a0.z, a0.w, a1.x, a1.y, a1.z, a1.w};
    u16 pk[8], pb[8];
#pragma unroll
    for (int j = 0; j < 8; ++j) {
      float nv = vals[j] * sc;
      pk[j] = f2bfu(nv);
      pb[j] = f2bfu(nv * bt);
    }
    u32x4 pkv = pack8(pk);
    *(u32x4*)(kn_g + (size_t)(bh * Ll + l0 + r) * 256 + lane * 8) = pkv;
    *(u32x4*)(&knb[r][lane * 8]) = pkv;
    *(u32x4*)(&kbb[r][lane * 8]) = pack8(pb);
#pragma unroll
    for (int j = 0; j < 8; ++j) {
      int d = lane * 8 + j;
      kbt[d][r] = pb[j];
      knT_g[(size_t)(bh * NCH + ch) * 8192 + (size_t)d * 32 + r] = pk[j];
    }
  }
#pragma unroll
  for (int rr = 0; rr < 4; ++rr) {   // v rows: (v*beta)^T
    int r = wv * 4 + rr;
    float bt = betas[r];
    const float* rowp = vc + (size_t)(b * Ll + l0 + r) * 1024 + h * 256;
    f32x4 a0 = *(const f32x4*)(rowp + lane * 8);
    f32x4 a1 = *(const f32x4*)(rowp + lane * 8 + 4);
    float vals[8] = {a0.x, a0.y, a0.z, a0.w, a1.x, a1.y, a1.z, a1.w};
#pragma unroll
    for (int j = 0; j < 8; ++j) vtb[lane * 8 + j][r] = f2bfu(vals[j] * bt);
  }
  __syncthreads();

  if (wv < 4) {  // A = tril(kb @ kn^T, -1)
    int mt = wv >> 1, nt = wv & 1;
    v8f acc = {};
#pragma unroll
    for (int kt = 0; kt < 8; ++kt) {
      v16bf a = ldfragA(&kbb[mt * 16][0] + kt * 32, 256);
      v16bf bb = ldfragBT(&knb[nt * 16][0] + kt * 32, 256);
      acc = wmma_bf(a, bb, acc);
    }
    int n = (lane & 15) + nt * 16, mb = (lane >> 4) * 8 + mt * 16;
#pragma unroll
    for (int v = 0; v < 8; ++v) {
      int m = mb + v;
      Af[m][n] = (n < m) ? acc[v] : 0.f;
    }
  }
  __syncthreads();

  if (wv == 0) {  // T = inv(I + A): forward substitution, lane = column
    float t[32];
    int c = lane;
    for (int i = 0; i < 32; ++i) {
      float val = (i == c) ? 1.f : 0.f;
      for (int j = 0; j < i; ++j) val -= Af[i][j] * t[j];
      t[i] = val;
      Tb[i][c] = f2bfu(val);
    }
  }
  __syncthreads();

  {  // u = T @ (v*beta), w = T @ (k*beta): 64 tile-tasks / 8 waves
    u16* ug = u_g + (size_t)(bh * NCH + ch) * 8192;
    u16* wg = w_g + (size_t)(bh * NCH + ch) * 8192;
#pragma unroll
    for (int s = 0; s < 8; ++s) {
      int tt = wv * 8 + s;
      int isw = tt >> 5;
      int t5 = tt & 31;
      int mt = t5 & 1, nt = t5 >> 1;
      v16bf a = ldfragA(&Tb[mt * 16][0], 32);
      v16bf bb = ldfragBT(isw ? &kbt[nt * 16][0] : &vtb[nt * 16][0], 32);
      v8f z = {};
      v8f acc = wmma_bf(a, bb, z);
      u16* dst = isw ? wg : ug;
      int n = (lane & 15) + nt * 16, mb = (lane >> 4) * 8 + mt * 16;
#pragma unroll
      for (int v = 0; v < 8; ++v) dst[(size_t)(mb + v) * 256 + n] = f2bfu(acc[v]);
    }
  }
}

// ---------------------------------------------------------------------------
// Delta-rule scan: grid = (b,h,dv_block) = 64 blocks; S[:,blk] lives in LDS.
// Per chunk: u2 = u - w*S ; attn = tril(q k^T) ; o = q*S + attn*u2 ; S += k^T*u2.
// ---------------------------------------------------------------------------
__global__ __launch_bounds__(256) void delta_scan(
    const u16* __restrict__ qn_g, const u16* __restrict__ kn_g,
    const u16* __restrict__ knT_g, const u16* __restrict__ u_g,
    const u16* __restrict__ w_g, float* __restrict__ dout) {
  __shared__ __align__(16) float Sf[256][32];
  __shared__ __align__(16) u16 SbT[32][256];   // SbT[n][k] = bf16(S[k][blk0+n])
  __shared__ __align__(16) u16 attnb[32][32];
  __shared__ __align__(16) u16 u2T[32][32];    // u2T[n][m] = bf16(u2[m][n])

  int blk = blockIdx.x & 7;
  int h = (blockIdx.x >> 3) & 3;
  int b = blockIdx.x >> 5;
  int bh = b * 4 + h;
  int blk0 = blk * 32;
  int tid = threadIdx.x, wv = tid >> 5, lane = tid & 31;

  for (int i = tid; i < 256 * 32; i += 256) (&Sf[0][0])[i] = 0.f;
  for (int i = tid; i < 32 * 256; i += 256) (&SbT[0][0])[i] = 0;
  __syncthreads();

  for (int ch = 0; ch < NCH; ++ch) {
    const u16* qn = qn_g + (size_t)(bh * Ll + ch * 32) * 256;
    const u16* kn = kn_g + (size_t)(bh * Ll + ch * 32) * 256;
    const u16* kT = knT_g + (size_t)(bh * NCH + ch) * 8192;
    const u16* uu = u_g + (size_t)(bh * NCH + ch) * 8192;
    const u16* ww = w_g + (size_t)(bh * NCH + ch) * 8192;

    // Phase 1: waves 0-3: u2 = u - w@S ; waves 4-7: attn = tril(q k^T)
    if (wv < 4) {
      int mt = wv >> 1, nt = wv & 1;
      v8f acc = {};
#pragma unroll
      for (int kt = 0; kt < 8; ++kt) {
        v16bf a = ldfragA(ww + mt * 16 * 256 + kt * 32, 256);
        v16bf bb = ldfragBT(&SbT[nt * 16][0] + kt * 32, 256);
        acc = wmma_bf(a, bb, acc);
      }
      int n = (lane & 15) + nt * 16, mb = (lane >> 4) * 8 + mt * 16;
#pragma unroll
      for (int v = 0; v < 8; ++v) {
        int m = mb + v;
        float uv = bf2f(uu[(size_t)m * 256 + blk0 + n]);
        u2T[n][m] = f2bfu(uv - acc[v]);
      }
    } else {
      int w4 = wv - 4;
      int mt = w4 >> 1, nt = w4 & 1;
      v8f acc = {};
#pragma unroll
      for (int kt = 0; kt < 8; ++kt) {
        v16bf a = ldfragA(qn + mt * 16 * 256 + kt * 32, 256);
        v16bf bb = ldfragBT(kn + nt * 16 * 256 + kt * 32, 256);
        acc = wmma_bf(a, bb, acc);
      }
      int n = (lane & 15) + nt * 16, mb = (lane >> 4) * 8 + mt * 16;
#pragma unroll
      for (int v = 0; v < 8; ++v) {
        int m = mb + v;
        attnb[m][n] = f2bfu((n <= m) ? acc[v] : 0.f);
      }
    }
    __syncthreads();

    // Phase 2: waves 0-3: o = q@S + attn@u2 -> global ; waves 4-7: dS = k^T@u2
    v8f dacc[8];
    if (wv < 4) {
      int mt = wv >> 1, nt = wv & 1;
      v8f acc = {};
#pragma unroll
      for (int kt = 0; kt < 8; ++kt) {
        v16bf a = ldfragA(qn + mt * 16 * 256 + kt * 32, 256);
        v16bf bb = ldfragBT(&SbT[nt * 16][0] + kt * 32, 256);
        acc = wmma_bf(a, bb, acc);
      }
      v16bf a2 = ldfragA(&attnb[mt * 16][0], 32);
      v16bf b2 = ldfragBT(&u2T[nt * 16][0], 32);
      acc = wmma_bf(a2, b2, acc);
      int n = (lane & 15) + nt * 16, mb = (lane >> 4) * 8 + mt * 16;
#pragma unroll
      for (int v = 0; v < 8; ++v) {
        int m = mb + v;
        int l = ch * 32 + m;
        dout[((size_t)(b * Ll + l) * 4 + h) * 256 + blk0 + n] = acc[v];
      }
    } else {
      int w4 = wv - 4;
#pragma unroll
      for (int s = 0; s < 8; ++s) {
        int tt = w4 * 8 + s;          // 16 dk-tiles x 2 dv-tiles
        int mt = tt & 15, nt = tt >> 4;
        v16bf a = ldfragA(kT + mt * 16 * 32, 32);
        v16bf bb = ldfragBT(&u2T[nt * 16][0], 32);
        v8f z = {};
        dacc[s] = wmma_bf(a, bb, z);
      }
    }
    __syncthreads();

    // Phase 3: commit S += dS, refresh bf16 mirror
    if (wv >= 4) {
      int w4 = wv - 4;
#pragma unroll
      for (int s = 0; s < 8; ++s) {
        int tt = w4 * 8 + s;
        int mt = tt & 15, nt = tt >> 4;
        int n = (lane & 15) + nt * 16, mb = (lane >> 4) * 8 + mt * 16;
#pragma unroll
        for (int v = 0; v < 8; ++v) {
          int m = mb + v;
          float nsv = Sf[m][n] + dacc[s][v];
          Sf[m][n] = nsv;
          SbT[n][m] = f2bfu(nsv);
        }
      }
    }
    __syncthreads();
  }
}

// ---------------------------------------------------------------------------
// branch stats + gate input assembly: one block per (b,l)
// ---------------------------------------------------------------------------
__global__ __launch_bounds__(256) void stats_gate(
    const float* __restrict__ x, const float* __restrict__ fs,
    const float* __restrict__ fl, const float* __restrict__ dl,
    const float* __restrict__ vv, u16* __restrict__ gate) {
  int row = blockIdx.x, tid = threadIdx.x;
  int wv = tid >> 5, lane = tid & 31;
  u16* g = gate + (size_t)row * GIN;
  const float* xr = x + ((size_t)row << 10);
  for (int i = tid; i < 1024; i += 256) g[i] = f2bfu(xr[i]);
  const float* br[4] = {fs, fl, dl, vv};
#pragma unroll
  for (int rsel = 0; rsel < 2; ++rsel) {
    int rid = wv * 2 + rsel;           // 16 reductions: (branch, head)
    int branch = rid >> 2, head = rid & 3;
    const float* p = br[branch] + ((size_t)row << 10) + head * 256 + lane * 8;
    float s1 = 0.f, s2 = 0.f;
#pragma unroll
    for (int j = 0; j < 8; ++j) { float t = p[j]; s1 += t; s2 += t * t; }
#pragma unroll
    for (int m = 16; m; m >>= 1) { s1 += __shfl_xor(s1, m, 32); s2 += __shfl_xor(s2, m, 32); }
    if (lane == 0) {
      float mean = s1 * (1.f / 256.f);
      float var = s2 * (1.f / 256.f) - mean * mean;
      g[1024 + branch * 8 + head] = f2bfu(mean);
      g[1024 + branch * 8 + 4 + head] = f2bfu(sqrtf(fmaxf(var, 1e-6f)));
    }
  }
}

// ---------------------------------------------------------------------------
// softmax gate + branch mix + per-head RMSNorm -> o_bf. One block per (b,l).
// ---------------------------------------------------------------------------
__global__ __launch_bounds__(256) void gate_combine(
    const float* __restrict__ logits, const float* __restrict__ b2,
    const float* __restrict__ glt, const float* __restrict__ fs,
    const float* __restrict__ fl, const float* __restrict__ dl,
    const float* __restrict__ vv, const float* __restrict__ onw,
    u16* __restrict__ obf) {
  __shared__ float wsm[4][4];
  __shared__ float red[8][4];
  int row = blockIdx.x, tid = threadIdx.x;
  int wv = tid >> 5, lane = tid & 31;
  if (tid < 4) {
    float t = softplusf_(glt[tid]) + 1e-4f;
    float lg[4], mx = -1e30f;
#pragma unroll
    for (int j = 0; j < 4; ++j) {
      lg[j] = (logits[row * 16 + tid * 4 + j] + b2[tid * 4 + j]) / t;
      mx = fmaxf(mx, lg[j]);
    }
    float s = 0.f;
#pragma unroll
    for (int j = 0; j < 4; ++j) { lg[j] = expf(lg[j] - mx); s += lg[j]; }
#pragma unroll
    for (int j = 0; j < 4; ++j) wsm[tid][j] = lg[j] / s;
  }
  __syncthreads();
  size_t base = (size_t)row << 10;
  float val[4];
#pragma unroll
  for (int hh = 0; hh < 4; ++hh) {
    size_t c = base + hh * 256 + tid;
    val[hh] = wsm[hh][0] * fs[c] + wsm[hh][1] * fl[c] + wsm[hh][2] * dl[c] + wsm[hh][3] * vv[c];
    float s = val[hh] * val[hh];
#pragma unroll
    for (int m = 16; m; m >>= 1) s += __shfl_xor(s, m, 32);
    if (lane == 0) red[wv][hh] = s;
  }
  __syncthreads();
#pragma unroll
  for (int hh = 0; hh < 4; ++hh) {
    float ss = 0.f;
#pragma unroll
    for (int w2 = 0; w2 < 8; ++w2) ss += red[w2][hh];
    float r = rsqrtf(ss * (1.f / 256.f) + 1e-5f);
    obf[base + hh * 256 + tid] = f2bfu(val[hh] * r * onw[tid]);
  }
}

// ---------------------------------------------------------------------------
extern "C" void kernel_launch(void* const* d_in, const int* in_sizes, int n_in,
                              void* d_out, int out_size, void* d_ws, size_t ws_size,
                              hipStream_t stream) {
  (void)in_sizes; (void)n_in; (void)out_size; (void)ws_size;
  const float* x       = (const float*)d_in[0];
  const float* Wq      = (const float*)d_in[1];
  const float* Wk      = (const float*)d_in[2];
  const float* Wv      = (const float*)d_in[3];
  const float* Wb      = (const float*)d_in[4];
  const float* conv_qw = (const float*)d_in[5];
  const float* conv_kw = (const float*)d_in[6];
  const float* conv_vw = (const float*)d_in[7];
  const float* fir_sw  = (const float*)d_in[8];
  const float* fir_lw  = (const float*)d_in[9];
  const float* mlp_w1  = (const float*)d_in[10];
  const float* mlp_b1  = (const float*)d_in[11];
  const float* mlp_w2  = (const float*)d_in[12];
  const float* mlp_b2  = (const float*)d_in[13];
  const float* glt     = (const float*)d_in[14];
  const float* onw     = (const float*)d_in[15];
  const float* Wo      = (const float*)d_in[16];
  float* out = (float*)d_out;

  char* ws = (char*)d_ws;
  size_t off = 0;
  auto take = [&](size_t bytes) -> char* {
    char* p = ws + off;
    off += (bytes + 255) & ~(size_t)255;
    return p;
  };
  const size_t E = (size_t)BLn * 1024;                       // 4,194,304
  u16* x_bf   = (u16*)take(E * 2);
  u16* wqT    = (u16*)take((size_t)1024 * 1024 * 2);         // [N,K] bf16
  u16* wkT    = (u16*)take((size_t)1024 * 1024 * 2);
  u16* wvT    = (u16*)take((size_t)1024 * 1024 * 2);
  u16* w1T    = (u16*)take((size_t)MLPH * GIN * 2);          // [2048,1056]
  u16* w2T    = (u16*)take((size_t)16 * MLPH * 2);           // [16,2048]
  u16* woT    = (u16*)take((size_t)1024 * 1024 * 2);
  float* qp   = (float*)take(E * 4);
  float* kp   = (float*)take(E * 4);
  float* vp   = (float*)take(E * 4);
  float* qc   = (float*)take(E * 4);
  float* kc   = (float*)take(E * 4);
  float* vc   = (float*)take(E * 4);
  float* beta = (float*)take((size_t)BLn * 4 * 4);
  u16* qn     = (u16*)take(E * 2);     // [b,h,L,dk]
  u16* kn     = (u16*)take(E * 2);
  u16* knT    = (u16*)take(E * 2);     // per chunk [256][32]
  u16* ub     = (u16*)take(E * 2);     // u = T(v*beta), per chunk [32][256]
  u16* wb     = (u16*)take(E * 2);     // w = T(k*beta)
  float* dlt  = (float*)take(E * 4);   // delta_out [b,l,h,dv]
  float* fs   = (float*)take(E * 4);
  float* fl   = (float*)take(E * 4);
  u16* gate   = (u16*)take((size_t)BLn * GIN * 2);
  float* mlph = (float*)take((size_t)BLn * MLPH * 4);
  u16* h_bf   = (u16*)take((size_t)BLn * MLPH * 2);
  float* lgts = (float*)take((size_t)BLn * 16 * 4);
  u16* o_bf   = (u16*)take(E * 2);

  dim3 T(256);
  const int EB = (int)((E + 255) / 256);   // 16384

  // 1) precision conversions (weights transposed once to [N,K] for async GEMM)
  cvt_bf16<<<EB, T, 0, stream>>>(x, x_bf, (int)E);
  transpose_bf16<<<4096, T, 0, stream>>>(Wq, wqT, 1024, 1024);
  transpose_bf16<<<4096, T, 0, stream>>>(Wk, wkT, 1024, 1024);
  transpose_bf16<<<4096, T, 0, stream>>>(Wv, wvT, 1024, 1024);
  transpose_bf16<<<(GIN * MLPH + 255) / 256, T, 0, stream>>>(mlp_w1, w1T, GIN, MLPH);
  transpose_bf16<<<(MLPH * 16 + 255) / 256, T, 0, stream>>>(mlp_w2, w2T, MLPH, 16);
  transpose_bf16<<<4096, T, 0, stream>>>(Wo, woT, 1024, 1024);

  // 2) q/k/v projections (WMMA + async staging)
  gemm_bf16_async<<<dim3(1024 / 128, BLn / 64), T, 0, stream>>>(x_bf, wqT, qp, BLn, 1024, 1024);
  gemm_bf16_async<<<dim3(1024 / 128, BLn / 64), T, 0, stream>>>(x_bf, wkT, kp, BLn, 1024, 1024);
  gemm_bf16_async<<<dim3(1024 / 128, BLn / 64), T, 0, stream>>>(x_bf, wvT, vp, BLn, 1024, 1024);

  // 3) causal depthwise convs + silu; beta
  conv_dw<4, true><<<EB, T, 0, stream>>>(qp, conv_qw, qc, (int)E);
  conv_dw<4, true><<<EB, T, 0, stream>>>(kp, conv_kw, kc, (int)E);
  conv_dw<4, true><<<EB, T, 0, stream>>>(vp, conv_vw, vc, (int)E);
  beta_k<<<(BLn * 4) / 256, T, 0, stream>>>(x, Wb, beta);

  // 4) delta rule: parallel prep, then chunkwise scan (WMMA, LDS-resident S)
  delta_prep<<<Bb * NHh * NCH, T, 0, stream>>>(qc, kc, vc, beta, qn, kn, knT, ub, wb);
  delta_scan<<<Bb * NHh * (DVv / 32), T, 0, stream>>>(qn, kn, knT, ub, wb, dlt);

  // 5) FIR branches
  conv_dw<3, false><<<EB, T, 0, stream>>>(vc, fir_sw, fs, (int)E);
  conv_dw<63, false><<<EB, T, 0, stream>>>(vc, fir_lw, fl, (int)E);

  // 6) gate input, MLP (WMMA), softmax gate + mix + RMSNorm
  stats_gate<<<BLn, T, 0, stream>>>(x, fs, fl, dlt, vc, gate);
  gemm_bf16_async<<<dim3(MLPH / 128, BLn / 64), T, 0, stream>>>(gate, w1T, mlph, BLn, MLPH, GIN);
  gelu_bias<<<(BLn * MLPH + 255) / 256, T, 0, stream>>>(mlph, mlp_b1, h_bf, BLn * MLPH);
  gemm_n16<<<BLn / 128, T, 0, stream>>>(h_bf, w2T, lgts, MLPH);
  gate_combine<<<BLn, T, 0, stream>>>(lgts, mlp_b2, glt, fs, fl, dlt, vc, onw, o_bf);

  // 7) output projection (WMMA) straight into d_out
  gemm_bf16_async<<<dim3(1024 / 128, BLn / 64), T, 0, stream>>>(o_bf, woT, out, BLn, 1024, 1024);
}
